// Decoder_12558484373770
// MI455X (gfx1250) — matrix operand order repaired and checked by reading, not compiled, not run on hardware
//
#include <hip/hip_runtime.h>
#include <hip/hip_bf16.h>
#include <stdint.h>

#define BATCH 128
#define WIN   64
#define EH    256   // E (enc hidden)
#define DH    256   // D (dec hidden)
#define FO    4     // F (out feats)

typedef __attribute__((ext_vector_type(16))) __bf16 v16bf;
typedef __attribute__((ext_vector_type(8)))  float  v8f;

__device__ __forceinline__ __bf16 f2bf(float f) {
  union { float f; uint32_t u; } v; v.f = f;
  uint32_t r = v.u + 0x7FFFu + ((v.u >> 16) & 1u);   // round-to-nearest-even
  uint16_t h = (uint16_t)(r >> 16);
  __bf16 o; __builtin_memcpy(&o, &h, 2);
  return o;
}

// packed f32x2 -> bf16x2 (1 VALU op on gfx1250: v_cvt_pk_bf16_f32)
__device__ __forceinline__ uint32_t f2bf2(float a, float b) {
#if __has_builtin(__builtin_amdgcn_cvt_pk_bf16_f32)
  auto pk = __builtin_amdgcn_cvt_pk_bf16_f32(a, b);   // .l = a, .h = b
  uint32_t r = 0;
  __builtin_memcpy(&r, &pk, sizeof(pk) < 4 ? sizeof(pk) : 4);
  return r;
#else
  union { __bf16 h[2]; uint32_t u; } x;
  x.h[0] = f2bf(a); x.h[1] = f2bf(b);
  return x.u;
#endif
}

__device__ __forceinline__ float fast_tanh(float x) {
#if __has_builtin(__builtin_amdgcn_tanhf)
  return __builtin_amdgcn_tanhf(x);        // gfx1250 v_tanh_f32
#elif __has_builtin(__builtin_amdgcn_tanh_f32)
  return __builtin_amdgcn_tanh_f32(x);
#else
  return tanhf(x);
#endif
}

__device__ __forceinline__ float sigm(float x) {
  return 1.0f / (1.0f + __expf(-x));
}

// ---- WMMA fragment loaders (wave32, contiguous-k order shared by A and B) -------------------
// A: 16x32 bf16, row-major source (LDS). lanes 0-15: M=lane, K 0..15;
// lanes 16-31: M=lane-16, K 16..31 (contiguous, matches B).
__device__ __forceinline__ v16bf load_a_frag(const __bf16* A, int lda, int kbase) {
  const int lane = threadIdx.x & 31;
  const int m  = lane & 15;
  const int hl = lane >> 4;
  const __bf16* row = A + m * lda + kbase + hl * 16;
  union { v16bf v; uint32_t d[8]; } u;
#pragma unroll
  for (int i = 0; i < 8; ++i) {
    u.d[i]     = *(const uint32_t*)(row + 2 * i);        // K = hl*16 + 2i,2i+1
  }
  return u.v;
}

// B: 32x16 bf16 where B[k][n] = W[n][k] (W row-major [N][K] prepacked bf16).
// lane (n = lane&15, hl = lane>>4) holds K = hl*16 + e, e=0..15 (contiguous).
__device__ __forceinline__ v16bf load_b_frag_bf16(const __bf16* W, int ldw, int n0, int kbase) {
  const int lane = threadIdx.x & 31;
  const __bf16* p = W + (size_t)(n0 + (lane & 15)) * ldw + kbase + (lane >> 4) * 16;
  union { v16bf v; uint4 q[2]; } u;
  u.q[0] = *(const uint4*)(p);
  u.q[1] = *(const uint4*)(p + 8);
  return u.v;
}

// ---- kernel 0: prepack attn_w1 (all 768 cols) and w_hh to bf16 ------------
// pair-granular: 256*768/2 + 1024*256/2 = 229376 pairs
__global__ void __launch_bounds__(256)
prepack_kernel(const float* __restrict__ attn_w1, const float* __restrict__ w_hh,
               uint32_t* __restrict__ w1bf2, uint32_t* __restrict__ whh2) {
  int i = blockIdx.x * 256 + threadIdx.x;
  if (i < 256 * 768 / 2) {
    w1bf2[i] = f2bf2(attn_w1[2 * i], attn_w1[2 * i + 1]);
  } else {
    int j = i - 256 * 768 / 2;
    if (j < 1024 * 256 / 2) whh2[j] = f2bf2(w_hh[2 * j], w_hh[2 * j + 1]);
  }
}

// ---- kernel 1: Ge = enc @ W1e^T   (M=8192, N=256, K=256), no bias ---------
__global__ void __launch_bounds__(512)
ge_gemm_kernel(const float* __restrict__ enc, const __bf16* __restrict__ w1bf,
               float* __restrict__ Ge) {
  __shared__ __align__(16) __bf16 a_s[16][256];
  const int tid = threadIdx.x;
  const int m0 = blockIdx.x * 16;
  uint32_t* a32 = (uint32_t*)&a_s[0][0];
  for (int i = tid; i < 16 * 128; i += 512) {          // pairs
    int r = i >> 7, kp = i & 127;
    const float* src = enc + (size_t)(m0 + r) * 256 + kp * 2;
    a32[r * 128 + kp] = f2bf2(src[0], src[1]);
  }
  __syncthreads();
  const int wave = tid >> 5, lane = tid & 31;
  const int n0 = wave * 16;
  const int nn = n0 + (lane & 15);
  const int hl = lane >> 4;
  v8f acc = {0.f, 0.f, 0.f, 0.f, 0.f, 0.f, 0.f, 0.f};
#pragma unroll
  for (int ks = 0; ks < 8; ++ks) {
    v16bf a = load_a_frag(&a_s[0][0], 256, ks * 32);
    v16bf b = load_b_frag_bf16(w1bf, 768, n0, 512 + ks * 32);  // W1e cols
    acc = __builtin_amdgcn_wmma_f32_16x16x32_bf16(false, a, false, b,
                                                  (short)0, acc, false, false);
  }
#pragma unroll
  for (int r = 0; r < 8; ++r)
    Ge[(size_t)(m0 + r + 8 * hl) * 256 + nn] = acc[r];
}

// ---- kernel 2: persistent 64-step scan, 16 batch rows per block -----------
__global__ void __launch_bounds__(512)
decoder_scan_kernel(const float* __restrict__ enc, const float* __restrict__ yhist,
                    const float* __restrict__ Ge,
                    const __bf16* __restrict__ w1bf, const __bf16* __restrict__ whh,
                    const float* __restrict__ attn_b1, const float* __restrict__ attn_w2,
                    const float* __restrict__ attn_b2,
                    const float* __restrict__ w_ih, const float* __restrict__ b_ih,
                    const float* __restrict__ b_hh,
                    const float* __restrict__ fc_w, const float* __restrict__ fc_b,
                    const float* __restrict__ fcf_w, const float* __restrict__ fcf_b,
                    float* __restrict__ out) {
  __shared__ float h_s[16][256];
  __shared__ float c_s[16][256];
  __shared__ __align__(16) __bf16 hc_bf[16][512];   // [h | c] bf16, A operand
  __shared__ float hterm[16][256];
  __shared__ float gates[16][1024];
  __shared__ float attn_sm[16][64];
  __shared__ float ctx_s[16][256];
  __shared__ float ytl[16][4];

  const int tid  = threadIdx.x;
  const int wave = tid >> 5;
  const int lane = tid & 31;
  const int hl   = lane >> 4;
  const int l16  = lane & 15;
  const int b0   = blockIdx.x * 16;

  for (int i = tid; i < 16 * 256; i += 512) {
    ((float*)h_s)[i] = 0.f; ((float*)c_s)[i] = 0.f; ((float*)ctx_s)[i] = 0.f;
  }
  __syncthreads();

  for (int t = 0; t < WIN; ++t) {
    // 1. pack [h|c] -> bf16 (packed pair converts)
    {
      uint32_t* hc32 = (uint32_t*)&hc_bf[0][0];
      for (int i = tid; i < 16 * 256; i += 512) {     // 4096 pairs
        int m = i >> 8, kp = i & 255;
        int k = kp * 2;
        float a, b;
        if (k < 256) { a = h_s[m][k];       b = h_s[m][k + 1]; }
        else         { a = c_s[m][k - 256]; b = c_s[m][k - 255]; }
        hc32[m * 256 + kp] = f2bf2(a, b);
      }
    }
    __syncthreads();

    // 2. hterm = [h|c] @ W1hc^T + b1   (M16,N256,K512); 1 n-tile per wave
    {
      const int n0 = wave * 16;
      const float bias = attn_b1[n0 + l16];
      v8f acc = {bias, bias, bias, bias, bias, bias, bias, bias};
#pragma unroll
      for (int ks = 0; ks < 16; ++ks) {
        v16bf a = load_a_frag(&hc_bf[0][0], 512, ks * 32);
        v16bf b = load_b_frag_bf16(w1bf, 768, n0, ks * 32);   // W1h|W1c cols
        acc = __builtin_amdgcn_wmma_f32_16x16x32_bf16(false, a, false, b,
                                                      (short)0, acc, false, false);
      }
#pragma unroll
      for (int r = 0; r < 8; ++r) hterm[r + 8 * hl][n0 + l16] = acc[r];
    }
    __syncthreads();

    // 3. scores[m][w] = b2 + sum_e w2[e]*tanh(Ge[m,w,e] + hterm[m,e])
    for (int p = tid; p < 16 * 64; p += 512) {
      int m = p >> 6, w = p & 63;
      const float* ge_row = Ge + ((size_t)(b0 + m) * WIN + w) * EH;
      int pn = p + 512;                                 // prefetch next row
      if (pn < 16 * 64) {
        const float* nxt = Ge + ((size_t)(b0 + (pn >> 6)) * WIN + (pn & 63)) * EH;
        __builtin_prefetch(nxt, 0, 1);                  // global_prefetch_b8
      }
      float s = attn_b2[0];
#pragma unroll 4
      for (int e = 0; e < EH; ++e)
        s += attn_w2[e] * fast_tanh(ge_row[e] + hterm[m][e]);
      attn_sm[m][w] = s;
    }
    __syncthreads();

    // 4. softmax over w (wave m owns row m; 2 elems/lane + shfl reduce)
    {
      const int m = wave;
      float v0 = attn_sm[m][lane], v1 = attn_sm[m][lane + 32];
      float mx = fmaxf(v0, v1);
#pragma unroll
      for (int off = 16; off > 0; off >>= 1) mx = fmaxf(mx, __shfl_xor(mx, off, 32));
      float e0 = __expf(v0 - mx), e1 = __expf(v1 - mx);
      float sm = e0 + e1;
#pragma unroll
      for (int off = 16; off > 0; off >>= 1) sm += __shfl_xor(sm, off, 32);
      float inv = 1.0f / sm;
      attn_sm[m][lane] = e0 * inv;
      attn_sm[m][lane + 32] = e1 * inv;
    }
    __syncthreads();

    // 5. context[m][e] = sum_w attn[m][w] * enc[b0+m][w][e]
    for (int i = tid; i < 16 * 256; i += 512) {
      int m = i >> 8, e = i & 255;
      const float* erow = enc + (size_t)(b0 + m) * WIN * EH + e;
      float s = 0.f;
#pragma unroll 8
      for (int w = 0; w < WIN; ++w) s += attn_sm[m][w] * erow[w * EH];
      ctx_s[m][e] = s;
    }
    __syncthreads();

    // 6. y_tilde = [ctx | y_t] @ fc_w^T + fc_b   (16x4, tiny)
    if (tid < 64) {
      int m = tid >> 2, f = tid & 3;
      const float* wrow = fc_w + f * (EH + FO);
      float s = fc_b[f];
      for (int k = 0; k < EH; ++k) s += ctx_s[m][k] * wrow[k];
      const float* yrow = yhist + ((size_t)(b0 + m) * WIN + t) * FO;
#pragma unroll
      for (int j = 0; j < FO; ++j) s += yrow[j] * wrow[EH + j];
      ytl[m][f] = s;
    }
    __syncthreads();

    // 7. gates = ytl@w_ih^T + b_ih + b_hh + h@w_hh^T  (M16,N1024,K256)
    //    A fragments hoisted once, reused across the wave's 4 n-tiles.
    {
      v16bf afr[8];
#pragma unroll
      for (int ks = 0; ks < 8; ++ks)
        afr[ks] = load_a_frag(&hc_bf[0][0], 512, ks * 32);   // h = cols 0..255
#pragma unroll
      for (int q = 0; q < 4; ++q) {
        const int n0 = (wave * 4 + q) * 16;
        const int nn = n0 + l16;
        const float base = b_ih[nn] + b_hh[nn];
        v8f acc;
#pragma unroll
        for (int r = 0; r < 8; ++r) {
          int m = r + 8 * hl;
          acc[r] = base + ytl[m][0] * w_ih[nn * 4 + 0] + ytl[m][1] * w_ih[nn * 4 + 1]
                        + ytl[m][2] * w_ih[nn * 4 + 2] + ytl[m][3] * w_ih[nn * 4 + 3];
        }
#pragma unroll
        for (int ks = 0; ks < 8; ++ks) {
          v16bf b = load_b_frag_bf16(whh, 256, n0, ks * 32);
          acc = __builtin_amdgcn_wmma_f32_16x16x32_bf16(false, afr[ks], false, b,
                                                        (short)0, acc, false, false);
        }
#pragma unroll
        for (int r = 0; r < 8; ++r) gates[r + 8 * hl][nn] = acc[r];
      }
    }
    __syncthreads();

    // 8. LSTM elementwise (torch gate order i,f,g,o)
    for (int i = tid; i < 16 * 256; i += 512) {
      int m = i >> 8, d = i & 255;
      float ig = sigm(gates[m][d]);
      float fg = sigm(gates[m][256 + d]);
      float gg = fast_tanh(gates[m][512 + d]);
      float og = sigm(gates[m][768 + d]);
      float cn = fg * c_s[m][d] + ig * gg;
      c_s[m][d] = cn;
      h_s[m][d] = og * fast_tanh(cn);
    }
    __syncthreads();
  }

  // final: out = [h | ctx] @ fcf_w^T + fcf_b
  if (tid < 64) {
    int m = tid >> 2, f = tid & 3;
    const float* wrow = fcf_w + f * (DH + EH);
    float s = fcf_b[f];
    for (int k = 0; k < DH; ++k) s += h_s[m][k] * wrow[k];
    for (int k = 0; k < EH; ++k) s += ctx_s[m][k] * wrow[DH + k];
    out[(b0 + m) * FO + f] = s;
  }
}

extern "C" void kernel_launch(void* const* d_in, const int* in_sizes, int n_in,
                              void* d_out, int out_size, void* d_ws, size_t ws_size,
                              hipStream_t stream) {
  const float* enc     = (const float*)d_in[0];
  const float* yhist   = (const float*)d_in[1];
  const float* attn_w1 = (const float*)d_in[2];
  const float* attn_b1 = (const float*)d_in[3];
  const float* attn_w2 = (const float*)d_in[4];
  const float* attn_b2 = (const float*)d_in[5];
  const float* w_ih    = (const float*)d_in[6];
  const float* w_hh    = (const float*)d_in[7];
  const float* b_ih    = (const float*)d_in[8];
  const float* b_hh    = (const float*)d_in[9];
  const float* fc_w    = (const float*)d_in[10];
  const float* fc_b    = (const float*)d_in[11];
  const float* fcf_w   = (const float*)d_in[12];
  const float* fcf_b   = (const float*)d_in[13];

  char* ws = (char*)d_ws;
  const size_t ge_bytes   = (size_t)BATCH * WIN * EH * sizeof(float);   // 8 MB
  const size_t w1bf_bytes = (size_t)256 * 768 * sizeof(uint16_t);       // 384 KB
  float*  Ge   = (float*)ws;
  __bf16* w1bf = (__bf16*)(ws + ge_bytes);
  __bf16* whh  = (__bf16*)(ws + ge_bytes + w1bf_bytes);

  // pairs: 256*768/2 + 1024*256/2 = 229376 -> 896 blocks of 256
  prepack_kernel<<<896, 256, 0, stream>>>(attn_w1, w_hh,
                                          (uint32_t*)w1bf, (uint32_t*)whh);
  ge_gemm_kernel<<<(BATCH * WIN) / 16, 512, 0, stream>>>(enc, w1bf, Ge);
  decoder_scan_kernel<<<BATCH / 16, 512, 0, stream>>>(enc, yhist, Ge, w1bf, whh,
      attn_b1, attn_w2, attn_b2, w_ih, b_ih, b_hh, fc_w, fc_b, fcf_w, fcf_b,
      (float*)d_out);
}
